// BasicBlock_Q_13477607375131
// MI455X (gfx1250) — compile-verified, matrix operand order checked by reading, not run on hardware
//
#include <hip/hip_runtime.h>
#include <hip/hip_bf16.h>

typedef _Float16 h8   __attribute__((ext_vector_type(8)));
typedef _Float16 v16h __attribute__((ext_vector_type(16)));
typedef float    v8f  __attribute__((ext_vector_type(8)));
typedef float    v4f  __attribute__((ext_vector_type(4)));

#define CCH   64          // channels
#define HH    112
#define WW    112
#define NN    32
#define HWSZ  (HH*WW)                 // 12544 (multiple of 32)
#define NPIX  (NN*HWSZ)               // 401408
#define WELEMS (CCH*CCH*9)            // 36864

// padded NHWC layout: (H+2) x (W+2) per image, zero halo
#define PW    114
#define PH    114
#define PHWSZ (PH*PW)                 // 12996
#define PPIX  (NN*PHWSZ)              // 415872
#define PADT_BYTES ((size_t)PPIX * CCH * 2)   // 53,231,616

// LDS layout (halves): [0, 36864) = all 9 weight taps; then 8 waves x 1024 repack tiles
#define WSM_HALVES   WELEMS                   // 36864 halves = 73728 B
#define TSM_HALVES   (8 * 16 * CCH)           // 8192 halves = 16384 B
#define LDS_BYTES    ((WSM_HALVES + TSM_HALVES) * 2)   // 90112 B

// ---------------- workspace layout (bytes) ----------------
#define MAX1_OFF 0
#define MAX2_OFF 128
#define WQ1_OFF  256
#define WQ2_OFF  (WQ1_OFF + WELEMS*2)          // 256 + 73728
#define XH_OFF   (WQ2_OFF + WELEMS*2)          // 147712
#define A1_OFF   (XH_OFF + PADT_BYTES)

// ---------------- weight abs-max(tanh) reduction ----------------
__global__ __launch_bounds__(256) void absmax_tanh_kernel(const float* __restrict__ w,
                                                          int n, float* __restrict__ outp) {
    __shared__ float red[256];
    float m = 0.f;
    for (int i = threadIdx.x; i < n; i += 256) m = fmaxf(m, fabsf(tanhf(w[i])));
    red[threadIdx.x] = m;
    __syncthreads();
    for (int s = 128; s > 0; s >>= 1) {
        if (threadIdx.x < (unsigned)s) red[threadIdx.x] = fmaxf(red[threadIdx.x], red[threadIdx.x + s]);
        __syncthreads();
    }
    if (threadIdx.x == 0) outp[0] = red[0];
}

// ---------------- DoReFa 4-bit weight quantize + relayout [tap][cout][cin] f16 ----
__global__ __launch_bounds__(256) void quant_w_kernel(const float* __restrict__ w,
                                                      const float* __restrict__ maxp,
                                                      _Float16* __restrict__ wq) {
    int idx = blockIdx.x * 256 + threadIdx.x;     // over OIHW flat
    if (idx >= WELEMS) return;
    int kw = idx % 3;
    int kh = (idx / 3) % 3;
    int i  = (idx / 9) % CCH;
    int o  = idx / (9 * CCH);
    float t  = tanhf(w[idx]);
    float rm = 0.5f * __builtin_amdgcn_rcpf(maxp[0]);   // v_rcp_f32
    float tn = t * rm + 0.5f;
    float q  = rintf(tn * 15.f) * (1.f / 15.f);
    float val = 2.f * q - 1.f;
    int tap = kh * 3 + kw;
    wq[(tap * CCH + o) * CCH + i] = (_Float16)val;
}

// ---------------- NCHW f32 -> padded NHWC f16 (zero halo) ----------------
__global__ __launch_bounds__(256) void nchw_to_nhwc_pad_kernel(const float* __restrict__ x,
                                                               _Float16* __restrict__ xh) {
    int pp = blockIdx.x * 256 + threadIdx.x;      // padded pixel index
    if (pp >= PPIX) return;
    int n   = pp / PHWSZ;
    int rem = pp % PHWSZ;
    int ph  = rem / PW;
    int pw  = rem % PW;
    h8* dst = (h8*)(xh + (size_t)pp * CCH);
    if (ph == 0 || ph == PH - 1 || pw == 0 || pw == PW - 1) {
        h8 z = {};
#pragma unroll
        for (int j = 0; j < 8; ++j) dst[j] = z;
        return;
    }
    const float* src = x + (size_t)n * CCH * HWSZ + (size_t)(ph - 1) * WW + (pw - 1);
    _Float16 buf[CCH];
#pragma unroll
    for (int c = 0; c < CCH; ++c) buf[c] = (_Float16)src[(size_t)c * HWSZ];
#pragma unroll
    for (int j = 0; j < 8; ++j) dst[j] = *(h8*)(buf + j * 8);
}

// ---------------- zero the halo ring of a padded NHWC f16 tensor ----------------
__global__ __launch_bounds__(256) void halo_zero_kernel(_Float16* __restrict__ t) {
    int pp = blockIdx.x * 256 + threadIdx.x;
    if (pp >= PPIX) return;
    int rem = pp % PHWSZ;
    int ph  = rem / PW;
    int pw  = rem % PW;
    if (ph == 0 || ph == PH - 1 || pw == 0 || pw == PW - 1) {
        h8 z = {};
        h8* dst = (h8*)(t + (size_t)pp * CCH);
#pragma unroll
        for (int j = 0; j < 8; ++j) dst[j] = z;
    }
}

__device__ __forceinline__ float quant_act(float v) {
    v = fminf(fmaxf(v, 0.f), 1.f);
    return rintf(v * 15.f) * (1.f / 15.f);
}

// ---------------- fused implicit-GEMM conv3x3 + BN + quant-act ----------------
// Wave-level: 32 pixels (2 M-tiles) x 64 cout, 144 v_wmma per wave.
// Each LDS B-fragment feeds two independent WMMAs (one per M-tile), so the
// matrix pipe is not gated on dscnt every issue.
template <bool RES>
__global__ __launch_bounds__(256) void conv_bn_act_kernel(
    const _Float16* __restrict__ xin,   // padded NHWC f16
    const _Float16* __restrict__ wq,    // [9][64][64] f16
    const float* __restrict__ gamma, const float* __restrict__ beta,
    const float* __restrict__ mean,  const float* __restrict__ var,
    _Float16* __restrict__ actout,      // padded NHWC f16 (RES=false)
    const float* __restrict__ xres,     // NCHW f32 (RES=true)
    float* __restrict__ out)            // NCHW f32 (RES=true)
{
    extern __shared__ _Float16 smem[];  // [WSM_HALVES weights][8 x 1024 repack]

    const int tid  = threadIdx.x;
    const int lane = tid & 31;
    const int wave = tid >> 5;
    const int l15  = lane & 15;
    const int hi16 = (lane >> 4) & 1;

    // ---- stage ALL 9 weight taps into LDS once (4608 h8 chunks / 256 threads) ----
    {
        const h8* src = (const h8*)wq;
        h8* dst = (h8*)smem;
#pragma unroll
        for (int j = 0; j < 18; ++j) dst[tid + j * 256] = src[tid + j * 256];
    }

    const int Mbase = (blockIdx.x * 8 + wave) * 32;   // 32-pixel strip (one image)
    const int n     = Mbase / HWSZ;
    const int hwb   = Mbase % HWSZ;                   // strip base within image plane
    const int chan0 = hi16 * 8;                       // K sub-slice per lane half

    // per-M-tile padded pixel index and A base pointer
    int pp[2];
    const _Float16* apc[2];
#pragma unroll
    for (int mt = 0; mt < 2; ++mt) {
        const int hw = hwb + mt * 16 + l15;
        const int h  = hw / WW;
        const int w  = hw % WW;
        pp[mt]  = n * PHWSZ + (h + 1) * PW + (w + 1);
        apc[mt] = xin + (size_t)pp[mt] * CCH + chan0;
    }

    // per-lane BN params for the 4 cout tiles (cout = nt*16 + l15)
    // v_rsq_f32 approximation instead of IEEE div/sqrt expansion
    float inv[4], bia[4];
#pragma unroll
    for (int nt = 0; nt < 4; ++nt) {
        int c = nt * 16 + l15;
        float iv = gamma[c] * __builtin_amdgcn_rsqf(var[c] + 1e-5f);
        inv[nt] = iv;
        bia[nt] = beta[c] - mean[c] * iv;
    }

    // warm the residual rows while the matrix pipe works (global_prefetch_b8)
    if constexpr (RES) {
#pragma unroll
        for (int nt = 0; nt < 4; ++nt) {
            int c = nt * 16 + l15;
#pragma unroll
            for (int mt = 0; mt < 2; ++mt) {
                const float* rp = xres + ((size_t)n * CCH + c) * HWSZ + hwb + mt * 16 + hi16 * 8;
                __builtin_prefetch(rp, 0, 3);
            }
        }
    }

    __syncthreads();                                  // weights visible to all waves

    v8f acc[2][4] = {};

#pragma unroll
    for (int tap = 0; tap < 9; ++tap) {
        const int toff = ((tap / 3 - 1) * PW + (tap % 3 - 1)) * CCH;  // compile-time
        const _Float16* wsm = smem + tap * (CCH * CCH);

#pragma unroll
        for (int kc = 0; kc < 2; ++kc) {              // K = 64 cin in two chunks of 32
            union { v16h v; h8 h[2]; } a0, a1;
            a0.h[0] = *(const h8*)(apc[0] + toff + kc * 32);
            a0.h[1] = *(const h8*)(apc[0] + toff + kc * 32 + 16);
            a1.h[0] = *(const h8*)(apc[1] + toff + kc * 32);
            a1.h[1] = *(const h8*)(apc[1] + toff + kc * 32 + 16);
#pragma unroll
            for (int nt = 0; nt < 4; ++nt) {          // 4 x 16 cout tiles
                union { v16h v; h8 h[2]; } b;
                const _Float16* bp = wsm + (nt * 16 + l15) * CCH + kc * 32 + chan0;
                b.h[0] = *(const h8*)(bp);
                b.h[1] = *(const h8*)(bp + 16);
                acc[0][nt] = __builtin_amdgcn_wmma_f32_16x16x32_f16(
                    false, a0.v, false, b.v, (short)0, acc[0][nt], false, false);
                acc[1][nt] = __builtin_amdgcn_wmma_f32_16x16x32_f16(
                    false, a1.v, false, b.v, (short)0, acc[1][nt], false, false);
            }
        }
    }

    // ---- epilogue ----
    if constexpr (RES) {
        // lane holds fixed cout, 8 consecutive pixels -> vectorized NCHW b128 I/O
#pragma unroll
        for (int mt = 0; mt < 2; ++mt) {
#pragma unroll
            for (int nt = 0; nt < 4; ++nt) {
                const int c = nt * 16 + l15;
                const size_t base = ((size_t)n * CCH + c) * HWSZ + hwb + mt * 16 + hi16 * 8;
                v4f r0 = *(const v4f*)(xres + base);
                v4f r1 = *(const v4f*)(xres + base + 4);
                v4f o0, o1;
#pragma unroll
                for (int r = 0; r < 4; ++r) {
                    o0[r] = quant_act(acc[mt][nt][r]     * inv[nt] + bia[nt] + r0[r]);
                    o1[r] = quant_act(acc[mt][nt][r + 4] * inv[nt] + bia[nt] + r1[r]);
                }
                *(v4f*)(out + base)     = o0;
                *(v4f*)(out + base + 4) = o1;
            }
        }
    } else {
        // repack (pixel-major, c-contiguous) through wave-private LDS tile (reused
        // for both M-tiles; same-wave LDS ops are in-order), then coalesced b128 stores
        _Float16* tsm = smem + WSM_HALVES + wave * (16 * CCH);
#pragma unroll
        for (int mt = 0; mt < 2; ++mt) {
#pragma unroll
            for (int nt = 0; nt < 4; ++nt) {
                const int c = nt * 16 + l15;
#pragma unroll
                for (int r = 0; r < 8; ++r) {
                    float v = quant_act(acc[mt][nt][r] * inv[nt] + bia[nt]);
                    tsm[(r + hi16 * 8) * CCH + c] = (_Float16)v;   // ds_store_b16
                }
            }
            const h8* srcT = (const h8*)(tsm + l15 * CCH + hi16 * 32);
            h8* dstG = (h8*)(actout + (size_t)pp[mt] * CCH + hi16 * 32);
#pragma unroll
            for (int j = 0; j < 4; ++j) dstG[j] = srcT[j];
        }
    }
}

extern "C" void kernel_launch(void* const* d_in, const int* in_sizes, int n_in,
                              void* d_out, int out_size, void* d_ws, size_t ws_size,
                              hipStream_t stream) {
    (void)in_sizes; (void)n_in; (void)out_size; (void)ws_size;
    const float* x  = (const float*)d_in[0];
    const float* w1 = (const float*)d_in[1];
    const float* g1 = (const float*)d_in[2];
    const float* b1 = (const float*)d_in[3];
    const float* m1 = (const float*)d_in[4];
    const float* v1 = (const float*)d_in[5];
    const float* w2 = (const float*)d_in[6];
    const float* g2 = (const float*)d_in[7];
    const float* b2 = (const float*)d_in[8];
    const float* m2 = (const float*)d_in[9];
    const float* v2 = (const float*)d_in[10];
    float* out = (float*)d_out;

    char* ws = (char*)d_ws;
    float*    max1 = (float*)(ws + MAX1_OFF);
    float*    max2 = (float*)(ws + MAX2_OFF);
    _Float16* wq1  = (_Float16*)(ws + WQ1_OFF);
    _Float16* wq2  = (_Float16*)(ws + WQ2_OFF);
    _Float16* xh   = (_Float16*)(ws + XH_OFF);    // padded NHWC f16
    _Float16* act1 = (_Float16*)(ws + A1_OFF);    // padded NHWC f16

    const int pad_blocks = (PPIX + 255) / 256;    // 1625

    absmax_tanh_kernel<<<1, 256, 0, stream>>>(w1, WELEMS, max1);
    absmax_tanh_kernel<<<1, 256, 0, stream>>>(w2, WELEMS, max2);
    quant_w_kernel<<<WELEMS / 256, 256, 0, stream>>>(w1, max1, wq1);
    quant_w_kernel<<<WELEMS / 256, 256, 0, stream>>>(w2, max2, wq2);
    nchw_to_nhwc_pad_kernel<<<pad_blocks, 256, 0, stream>>>(x, xh);
    halo_zero_kernel<<<pad_blocks, 256, 0, stream>>>(act1);

    const int conv_blocks = NPIX / (8 * 32);  // 1568: 8 waves x 32 pixels per block
    conv_bn_act_kernel<false><<<conv_blocks, 256, LDS_BYTES, stream>>>(
        xh, wq1, g1, b1, m1, v1, act1, nullptr, nullptr);
    conv_bn_act_kernel<true><<<conv_blocks, 256, LDS_BYTES, stream>>>(
        act1, wq2, g2, b2, m2, v2, nullptr, x, out);
}
